// LiftSplatShoot_81252191306140
// MI455X (gfx1250) — compile-verified, matrix operand order checked by reading, not run on hardware
//
#include <hip/hip_runtime.h>
#include <hip/hip_bf16.h>
#include <stdint.h>

typedef uint32_t u32;
typedef uint16_t u16;
typedef __bf16 bf16;
typedef bf16  bf16x16 __attribute__((ext_vector_type(16)));
typedef float f32x8   __attribute__((ext_vector_type(8)));

#define HFD 32
#define WFD 88
#define NPIX (HFD*WFD)        // 2816
#define BATCH 4
#define PTOT (BATCH*NPIX)     // 11264
#define CINC 512
#define MIDC 128
#define DBINS 64
#define CCH 64
#define HBEV 200
#define WBEV 200
#define HWBEV (HBEV*WBEV)     // 40000
#define HPAD (HFD+2)          // 34
#define WPAD (WFD+2)          // 90

// ---------- helpers ----------
__device__ __forceinline__ u16 f2bf(float f) {
  u32 x = __float_as_uint(f);
  x += 0x7FFFu + ((x >> 16) & 1u);   // round-to-nearest-even
  return (u16)(x >> 16);
}

union FragAB { bf16x16 bf; u32 u[8]; };
union FragC  { f32x8 v; float f[8]; };
union PackU  { uint4 q; u16 s[8]; };

__device__ __forceinline__ void loadFrag(FragAB& fr, const u16* p) {
  const uint4 q0 = *reinterpret_cast<const uint4*>(p);
  const uint4 q1 = *reinterpret_cast<const uint4*>(p + 16);
  fr.u[0]=q0.x; fr.u[1]=q0.y; fr.u[2]=q0.z; fr.u[3]=q0.w;
  fr.u[4]=q1.x; fr.u[5]=q1.y; fr.u[6]=q1.z; fr.u[7]=q1.w;
}

// ---------- prep kernels ----------
__global__ __launch_bounds__(256) void zero_u32(u32* __restrict__ p, int n) {
  int i = blockIdx.x * 256 + threadIdx.x;
  if (i < n) p[i] = 0u;
}

// features (B,512,2816) f32 -> featpad (B,34,90,512) bf16 (zero halo pre-set)
__global__ __launch_bounds__(256) void cvt_feat_pad(
    const float* __restrict__ in, u16* __restrict__ out) {
  __shared__ u16 tile[32][33];
  const int b  = blockIdx.z;
  const int c0 = blockIdx.y * 32;
  const int p0 = blockIdx.x * 32;
  const int tx = threadIdx.x & 31;
  const int ty = threadIdx.x >> 5;          // 0..7
  #pragma unroll
  for (int i = 0; i < 4; ++i) {
    const int c = ty * 4 + i;               // coalesced over tx (pixels)
    tile[c][tx] = f2bf(in[((size_t)(b * CINC + c0 + c)) * NPIX + p0 + tx]);
  }
  __syncthreads();
  #pragma unroll
  for (int i = 0; i < 4; ++i) {
    const int pl = ty * 4 + i;
    const int p  = p0 + pl;
    const int h  = p / WFD;
    const int w  = p - h * WFD;
    out[((size_t)((b * HPAD + h + 1) * WPAD + w + 1)) * CINC + c0 + tx] = tile[tx][pl];
  }
}

// w1 (128,512,3,3) -> w1t (tap*16+kc, 128, 32) bf16
__global__ __launch_bounds__(256) void prep_w1(const float* __restrict__ w1,
                                               u16* __restrict__ w1t) {
  int i = blockIdx.x * 256 + threadIdx.x;            // 589824 total
  int kk = i & 31;
  int m  = (i >> 5) & 127;
  int kc = (i >> 12) & 15;
  int t9 = i >> 16;
  int ci = kc * 32 + kk;
  w1t[i] = f2bf(w1[((size_t)m * CINC + ci) * 9 + t9]);
}

// w2 (64,128) -> w2t (kc, 64, 32)
__global__ __launch_bounds__(256) void prep_w2(const float* __restrict__ w2,
                                               u16* __restrict__ w2t) {
  int i = blockIdx.x * 256 + threadIdx.x;            // 8192 total
  int kk = i & 31;
  int m  = (i >> 5) & 63;
  int kc = i >> 11;
  w2t[i] = f2bf(w2[(size_t)m * MIDC + kc * 32 + kk]);
}

// wc (64,512) -> wct (kc, 64, 32)
__global__ __launch_bounds__(256) void prep_wc(const float* __restrict__ wc,
                                               u16* __restrict__ wct) {
  int i = blockIdx.x * 256 + threadIdx.x;            // 32768 total
  int kk = i & 31;
  int m  = (i >> 5) & 63;
  int kc = i >> 11;
  wct[i] = f2bf(wc[(size_t)m * CINC + kc * 32 + kk]);
}

__global__ __launch_bounds__(256) void prep_bn(
    const float* __restrict__ b1, const float* __restrict__ g1,
    const float* __restrict__ be1, const float* __restrict__ m1,
    const float* __restrict__ v1,
    const float* __restrict__ bc, const float* __restrict__ gc,
    const float* __restrict__ bec, const float* __restrict__ mc,
    const float* __restrict__ vc,
    float* __restrict__ s1, float* __restrict__ t1,
    float* __restrict__ sc, float* __restrict__ tc) {
  int t = threadIdx.x;
  if (t < 128) {
    float s = g1[t] * rsqrtf(v1[t] + 1e-5f);
    s1[t] = s;
    t1[t] = (b1[t] - m1[t]) * s + be1[t];
  }
  if (t < 64) {
    float s = gc[t] * rsqrtf(vc[t] + 1e-5f);
    sc[t] = s;
    tc[t] = (bc[t] - mc[t]) * s + bec[t];
  }
}

// ---------- K1: conv3x3 (512->128) + BN + ReLU, barrier-free implicit GEMM ----
// Block: 256 threads = 8 waves. Block tile M=128 x N=64 pixels.
// Wave tile: 32(M) x 32(N) = 2x2 WMMA subtiles -> 4 v_wmma per K-step.
// Padded input buffer => no predication, no bounds math in the hot loop.
__global__ __launch_bounds__(256) void conv3x3_bn_relu(
    const u16* __restrict__ featpad,  // (B, 34, 90, 512) bf16, zero halo
    const u16* __restrict__ w1t,      // (144, 128, 32) bf16
    const float* __restrict__ s1, const float* __restrict__ t1,
    u16* __restrict__ hpm)            // (PTOT, 128) bf16 pixel-major
{
  const int tid  = threadIdx.x;
  const int wv   = tid >> 5;
  const int lane = tid & 31;
  const int hi   = lane >> 4;
  const int nl   = lane & 15;
  const int mg   = wv & 3;            // M rows [mg*32, mg*32+32)
  const int ng   = wv >> 2;           // pixel group of 32 within 64-tile
  const int blockBase = blockIdx.x * 64;

  int pin[2];
  const u16* pbase[2];
  #pragma unroll
  for (int ns = 0; ns < 2; ++ns) {
    const int p  = blockBase + ng * 32 + ns * 16 + nl;
    const int b  = p / NPIX;
    const int hw = p - b * NPIX;
    const int h  = hw / WFD;
    const int w  = hw - h * WFD;
    pin[ns]   = p;
    pbase[ns] = featpad
              + ((size_t)((b * HPAD + h + 1) * WPAD + (w + 1))) * CINC + 8 * hi;
  }

  FragC acc[2][2];
  #pragma unroll
  for (int ms = 0; ms < 2; ++ms)
    #pragma unroll
    for (int ns = 0; ns < 2; ++ns)
      #pragma unroll
      for (int r = 0; r < 8; ++r) acc[ms][ns].f[r] = 0.0f;

  const u16* aptr = w1t + ((size_t)(mg * 32 + nl)) * 32 + 8 * hi;

  for (int t9 = 0; t9 < 9; ++t9) {
    const int dy = t9 / 3 - 1;                 // 9x only; constant divisor
    const int dx = t9 - (t9 / 3) * 3 - 1;
    const u16* bptr0 = pbase[0] + (ptrdiff_t)(dy * WPAD + dx) * CINC;
    const u16* bptr1 = pbase[1] + (ptrdiff_t)(dy * WPAD + dx) * CINC;
    __builtin_prefetch(bptr0, 0, 1);           // global_prefetch_b8
    #pragma unroll 4
    for (int kc = 0; kc < 16; ++kc) {
      FragAB a0, a1, b0, b1;
      loadFrag(a0, aptr);                      // M subtile 0
      loadFrag(a1, aptr + 16 * 32);            // M subtile 1
      loadFrag(b0, bptr0);                     // N subtile 0
      loadFrag(b1, bptr1);                     // N subtile 1
      acc[0][0].v = __builtin_amdgcn_wmma_f32_16x16x32_bf16(
          false, a0.bf, false, b0.bf, (short)0, acc[0][0].v, false, false);
      acc[0][1].v = __builtin_amdgcn_wmma_f32_16x16x32_bf16(
          false, a0.bf, false, b1.bf, (short)0, acc[0][1].v, false, false);
      acc[1][0].v = __builtin_amdgcn_wmma_f32_16x16x32_bf16(
          false, a1.bf, false, b0.bf, (short)0, acc[1][0].v, false, false);
      acc[1][1].v = __builtin_amdgcn_wmma_f32_16x16x32_bf16(
          false, a1.bf, false, b1.bf, (short)0, acc[1][1].v, false, false);
      aptr  += MIDC * 32;                      // next K-chunk of weights
      bptr0 += 32;                             // next 32 input channels
      bptr1 += 32;
    }
  }

  #pragma unroll
  for (int ms = 0; ms < 2; ++ms) {
    const int m0 = mg * 32 + ms * 16 + 8 * hi;
    #pragma unroll
    for (int ns = 0; ns < 2; ++ns) {
      PackU pk;
      #pragma unroll
      for (int r = 0; r < 8; ++r) {
        float val = fmaxf(fmaf(acc[ms][ns].f[r], s1[m0 + r], t1[m0 + r]), 0.0f);
        pk.s[r] = f2bf(val);
      }
      *reinterpret_cast<uint4*>(hpm + (size_t)pin[ns] * MIDC + m0) = pk.q;
    }
  }
}

// ---------- K2: conv1x1 (128->64) + bias + softmax over D ----------
// Block: 128 threads = 4 waves, tile M=64 x N=64. One barrier (softmax).
__global__ __launch_bounds__(128) void depth_logits_softmax(
    const u16* __restrict__ hpm,      // (PTOT, 128) bf16
    const u16* __restrict__ w2t,      // (4, 64, 32) bf16
    const float* __restrict__ b2,
    float* __restrict__ dp)           // (PTOT, 64) f32
{
  __shared__ float ldsL[DBINS][65];
  const int tid  = threadIdx.x;
  const int wv   = tid >> 5;
  const int lane = tid & 31;
  const int hi   = lane >> 4;
  const int nl   = lane & 15;
  const int mg   = wv & 1;
  const int ng   = wv >> 1;
  const int blockBase = blockIdx.x * 64;

  int pin[2];
  const u16* bptr[2];
  #pragma unroll
  for (int ns = 0; ns < 2; ++ns) {
    pin[ns]  = blockBase + ng * 32 + ns * 16 + nl;
    bptr[ns] = hpm + (size_t)pin[ns] * MIDC + 8 * hi;
  }

  FragC acc[2][2];
  #pragma unroll
  for (int ms = 0; ms < 2; ++ms)
    #pragma unroll
    for (int ns = 0; ns < 2; ++ns)
      #pragma unroll
      for (int r = 0; r < 8; ++r) acc[ms][ns].f[r] = 0.0f;

  const u16* aptr = w2t + ((size_t)(mg * 32 + nl)) * 32 + 8 * hi;

  #pragma unroll
  for (int kc = 0; kc < 4; ++kc) {
    FragAB a0, a1, b0, b1;
    loadFrag(a0, aptr);
    loadFrag(a1, aptr + 16 * 32);
    loadFrag(b0, bptr[0]);
    loadFrag(b1, bptr[1]);
    acc[0][0].v = __builtin_amdgcn_wmma_f32_16x16x32_bf16(
        false, a0.bf, false, b0.bf, (short)0, acc[0][0].v, false, false);
    acc[0][1].v = __builtin_amdgcn_wmma_f32_16x16x32_bf16(
        false, a0.bf, false, b1.bf, (short)0, acc[0][1].v, false, false);
    acc[1][0].v = __builtin_amdgcn_wmma_f32_16x16x32_bf16(
        false, a1.bf, false, b0.bf, (short)0, acc[1][0].v, false, false);
    acc[1][1].v = __builtin_amdgcn_wmma_f32_16x16x32_bf16(
        false, a1.bf, false, b1.bf, (short)0, acc[1][1].v, false, false);
    aptr   += 64 * 32;
    bptr[0] += 32;
    bptr[1] += 32;
  }

  #pragma unroll
  for (int ms = 0; ms < 2; ++ms)
    #pragma unroll
    for (int ns = 0; ns < 2; ++ns) {
      const int col = ng * 32 + ns * 16 + nl;
      #pragma unroll
      for (int r = 0; r < 8; ++r) {
        const int d = mg * 32 + ms * 16 + 8 * hi + r;
        ldsL[d][col] = acc[ms][ns].f[r] + b2[d];
      }
    }
  __syncthreads();

  if (tid < 64) {
    const int p = blockBase + tid;
    float mx = -3.0e38f;
    for (int d = 0; d < DBINS; ++d) mx = fmaxf(mx, ldsL[d][tid]);
    float sum = 0.0f;
    for (int d = 0; d < DBINS; ++d) {
      float e = __expf(ldsL[d][tid] - mx);
      ldsL[d][tid] = e;
      sum += e;
    }
    const float inv = 1.0f / sum;
    for (int d = 0; d < DBINS; ++d)
      dp[(size_t)p * DBINS + d] = ldsL[d][tid] * inv;
  }
}

// ---------- K3: conv1x1 (512->64) + BN + ReLU -> feat (PTOT, 64) ----------
__global__ __launch_bounds__(128) void compress_bn_relu(
    const u16* __restrict__ featpad,  // (B, 34, 90, 512) bf16
    const u16* __restrict__ wct,      // (16, 64, 32) bf16
    const float* __restrict__ sc, const float* __restrict__ tc,
    float* __restrict__ featT)        // (PTOT, 64) f32
{
  const int tid  = threadIdx.x;
  const int wv   = tid >> 5;
  const int lane = tid & 31;
  const int hi   = lane >> 4;
  const int nl   = lane & 15;
  const int mg   = wv & 1;
  const int ng   = wv >> 1;
  const int blockBase = blockIdx.x * 64;

  int pin[2];
  const u16* bptr[2];
  #pragma unroll
  for (int ns = 0; ns < 2; ++ns) {
    const int p  = blockBase + ng * 32 + ns * 16 + nl;
    const int b  = p / NPIX;
    const int hw = p - b * NPIX;
    const int h  = hw / WFD;
    const int w  = hw - h * WFD;
    pin[ns]  = p;
    bptr[ns] = featpad
             + ((size_t)((b * HPAD + h + 1) * WPAD + (w + 1))) * CINC + 8 * hi;
  }

  FragC acc[2][2];
  #pragma unroll
  for (int ms = 0; ms < 2; ++ms)
    #pragma unroll
    for (int ns = 0; ns < 2; ++ns)
      #pragma unroll
      for (int r = 0; r < 8; ++r) acc[ms][ns].f[r] = 0.0f;

  const u16* aptr = wct + ((size_t)(mg * 32 + nl)) * 32 + 8 * hi;

  #pragma unroll 4
  for (int kc = 0; kc < 16; ++kc) {
    FragAB a0, a1, b0, b1;
    loadFrag(a0, aptr);
    loadFrag(a1, aptr + 16 * 32);
    loadFrag(b0, bptr[0]);
    loadFrag(b1, bptr[1]);
    acc[0][0].v = __builtin_amdgcn_wmma_f32_16x16x32_bf16(
        false, a0.bf, false, b0.bf, (short)0, acc[0][0].v, false, false);
    acc[0][1].v = __builtin_amdgcn_wmma_f32_16x16x32_bf16(
        false, a0.bf, false, b1.bf, (short)0, acc[0][1].v, false, false);
    acc[1][0].v = __builtin_amdgcn_wmma_f32_16x16x32_bf16(
        false, a1.bf, false, b0.bf, (short)0, acc[1][0].v, false, false);
    acc[1][1].v = __builtin_amdgcn_wmma_f32_16x16x32_bf16(
        false, a1.bf, false, b1.bf, (short)0, acc[1][1].v, false, false);
    aptr   += 64 * 32;
    bptr[0] += 32;
    bptr[1] += 32;
  }

  #pragma unroll
  for (int ms = 0; ms < 2; ++ms) {
    const int m0 = mg * 32 + ms * 16 + 8 * hi;
    #pragma unroll
    for (int ns = 0; ns < 2; ++ns) {
      float vals[8];
      #pragma unroll
      for (int r = 0; r < 8; ++r)
        vals[r] = fmaxf(fmaf(acc[ms][ns].f[r], sc[m0 + r], tc[m0 + r]), 0.0f);
      float* dst = featT + (size_t)pin[ns] * CCH + m0;     // 32B-aligned
      *reinterpret_cast<float4*>(dst)     = make_float4(vals[0], vals[1], vals[2], vals[3]);
      *reinterpret_cast<float4*>(dst + 4) = make_float4(vals[4], vals[5], vals[6], vals[7]);
    }
  }
}

// ---------- K4: geometry + outer product + scatter-add ----------
__global__ __launch_bounds__(256) void splat(
    const float* __restrict__ dp,      // (PTOT, 64)
    const float* __restrict__ featT,   // (PTOT, 64)
    const float* __restrict__ intrin,  // (B,3,3)
    const float* __restrict__ l2c,     // (B,4,4)
    float* __restrict__ bevT)          // (B, 40000, 64)
{
  const int q = threadIdx.x >> 6;      // 4 (pixel,depth) pairs per block
  const int c = threadIdx.x & 63;
  const long long pair = (long long)blockIdx.x * 4 + q;
  const int d = (int)(pair & 63);
  const int p = (int)(pair >> 6);
  const int b  = p / NPIX;
  const int hw = p - b * NPIX;
  const int h  = hw / WFD;
  const int w  = hw - h * WFD;

  // analytic K^-1 ray (wave-uniform scalar math)
  const float* Km = intrin + b * 9;
  const float fx = Km[0], cx = Km[2], fy = Km[4], cy = Km[5];
  const float uu = (float)w * 32.0f;
  const float vv = (float)h * 32.0f;
  const float depthC = 1.0f + (74.0f / 64.0f) * ((float)d + 0.5f);
  const float Xc = (uu - cx) / fx * depthC;
  const float Yc = (vv - cy) / fy * depthC;
  const float Zc = depthC;

  // rigid inverse of lidar_to_camera: x_l = R^T (x_c - t)
  const float* T = l2c + b * 16;
  const float ex = Xc - T[3], ey = Yc - T[7], ez = Zc - T[11];
  const float xl = T[0] * ex + T[4] * ey + T[8] * ez;
  const float yl = T[1] * ex + T[5] * ey + T[9] * ez;

  const int bx = (int)((xl + 50.0f) * 2.0f);   // trunc toward zero == astype(int32)
  const int by = (int)((yl + 50.0f) * 2.0f);
  if ((unsigned)bx < WBEV && (unsigned)by < HBEV) {
    const float prob = dp[(size_t)p * DBINS + d];
    const float fv   = featT[(size_t)p * CCH + c];
    atomicAdd(&bevT[(((size_t)b * HWBEV) + (size_t)by * WBEV + bx) * CCH + c],
              prob * fv);
  }
}

// ---------- K5: (b, hw, c) -> (b, c, hw) ----------
__global__ __launch_bounds__(256) void bev_transpose(
    const float* __restrict__ bevT, float* __restrict__ out)
{
  __shared__ float tile[64][65];
  const int b = blockIdx.y;
  const int hwBase = blockIdx.x * 64;
  const int cIn = threadIdx.x & 63;
  const int g   = threadIdx.x >> 6;
  const float* src = bevT + ((size_t)b * HWBEV + hwBase) * CCH;
  #pragma unroll
  for (int i = 0; i < 16; ++i) {
    int row = g * 16 + i;
    tile[row][cIn] = src[(size_t)row * CCH + cIn];
  }
  __syncthreads();
  const int hwOut = threadIdx.x & 63;
  float* dst = out + (size_t)b * CCH * HWBEV + hwBase;
  #pragma unroll
  for (int i = 0; i < 16; ++i) {
    int cc = g * 16 + i;
    dst[(size_t)cc * HWBEV + hwOut] = tile[hwOut][cc];
  }
}

// ---------- launcher ----------
extern "C" void kernel_launch(void* const* d_in, const int* in_sizes, int n_in,
                              void* d_out, int out_size, void* d_ws, size_t ws_size,
                              hipStream_t stream) {
  (void)in_sizes; (void)n_in; (void)out_size; (void)ws_size;
  const float* features = (const float*)d_in[0];
  /* d_in[1] = depth (unused by reference) */
  const float* intrin = (const float*)d_in[2];
  const float* l2c    = (const float*)d_in[3];
  const float* w1  = (const float*)d_in[4];
  const float* b1  = (const float*)d_in[5];
  const float* g1  = (const float*)d_in[6];
  const float* be1 = (const float*)d_in[7];
  const float* m1  = (const float*)d_in[8];
  const float* v1  = (const float*)d_in[9];
  const float* w2  = (const float*)d_in[10];
  const float* b2  = (const float*)d_in[11];
  const float* wc  = (const float*)d_in[12];
  const float* bc  = (const float*)d_in[13];
  const float* gc  = (const float*)d_in[14];
  const float* bec = (const float*)d_in[15];
  const float* mc  = (const float*)d_in[16];
  const float* vc  = (const float*)d_in[17];
  float* out = (float*)d_out;

  char* ws = (char*)d_ws;
  size_t off = 0;
  auto take = [&](size_t bytes) -> char* {
    char* p = ws + off;
    off = (off + bytes + 255) & ~(size_t)255;
    return p;
  };
  u16*   featpad = (u16*) take((size_t)BATCH * HPAD * WPAD * CINC * 2); // 12.5 MB
  u16*   w1t    = (u16*)  take((size_t)144 * 128 * 32 * 2);             // 1.2 MB
  u16*   w2t    = (u16*)  take((size_t)4 * 64 * 32 * 2);
  u16*   wct    = (u16*)  take((size_t)16 * 64 * 32 * 2);
  float* s1     = (float*)take(128 * 4);
  float* t1     = (float*)take(128 * 4);
  float* sc     = (float*)take(64 * 4);
  float* tc     = (float*)take(64 * 4);
  u16*   hpm    = (u16*)  take((size_t)PTOT * MIDC * 2);                // 2.9 MB
  float* dp     = (float*)take((size_t)PTOT * DBINS * 4);               // 2.9 MB
  float* featT  = (float*)take((size_t)PTOT * CCH * 4);                 // 2.9 MB
  float* bevT   = (float*)take((size_t)BATCH * HWBEV * CCH * 4);        // 41 MB

  const int padDwords = BATCH * HPAD * WPAD * CINC / 2;                 // 3,133,440
  zero_u32<<<(padDwords + 255) / 256, 256, 0, stream>>>((u32*)featpad, padDwords);
  zero_u32<<<(BATCH * HWBEV * CCH + 255) / 256, 256, 0, stream>>>((u32*)bevT,
                                                                  BATCH * HWBEV * CCH);
  cvt_feat_pad<<<dim3(NPIX / 32, CINC / 32, BATCH), 256, 0, stream>>>(features, featpad);
  prep_w1<<<2304, 256, 0, stream>>>(w1, w1t);
  prep_w2<<<32, 256, 0, stream>>>(w2, w2t);
  prep_wc<<<128, 256, 0, stream>>>(wc, wct);
  prep_bn<<<1, 256, 0, stream>>>(b1, g1, be1, m1, v1, bc, gc, bec, mc, vc,
                                 s1, t1, sc, tc);
  conv3x3_bn_relu<<<PTOT / 64, 256, 0, stream>>>(featpad, w1t, s1, t1, hpm);
  depth_logits_softmax<<<PTOT / 64, 128, 0, stream>>>(hpm, w2t, b2, dp);
  compress_bn_relu<<<PTOT / 64, 128, 0, stream>>>(featpad, wct, sc, tc, featT);
  splat<<<(PTOT * DBINS) / 4, 256, 0, stream>>>(dp, featT, intrin, l2c, bevT);
  bev_transpose<<<dim3(HWBEV / 64, BATCH), 256, 0, stream>>>(bevT, out);
}
